// MultiHeadAttention_51694226375220
// MI455X (gfx1250) — compile-verified, matrix operand order checked by reading, not run on hardware
//
#include <hip/hip_runtime.h>

// Problem constants (match reference)
#define B_   2
#define S_   2048
#define D_   1024
#define H_   16
#define DK_  64

typedef __attribute__((ext_vector_type(16))) __bf16 v16bf;
typedef __attribute__((ext_vector_type(8)))  __bf16 v8bf;
typedef __attribute__((ext_vector_type(8)))  float  v8f;

union FragU { v16bf v; v8bf h[2]; };

__device__ __forceinline__ v8f wmma_bf16(v16bf a, v16bf b, v8f c) {
  // 8 args: (neg_a, A, neg_b, B, c_mod, C, reuse_a, reuse_b)
  return __builtin_amdgcn_wmma_f32_16x16x32_bf16(
      false, a, false, b, (short)0, c, false, false);
}

// 16-bit A/B fragment loader. ISA layout: per lane, elements 0..7 (VGPR0-3)
// come from K = khalf*8..+7, elements 8..15 (VGPR4-7) from K = 16+khalf*8..
// Caller passes p = row_base + khalf*8; second chunk is +16 elements.
__device__ __forceinline__ v16bf load_frag(const __bf16* p) {
  FragU u;
  u.h[0] = *(const v8bf*)(p);
  u.h[1] = *(const v8bf*)(p + 16);
  return u.v;
}

// Convert 8 consecutive f32 -> v8bf (half of an A fragment)
__device__ __forceinline__ v8bf cvt8(const float* p) {
  float4 x = *(const float4*)p;
  float4 y = *(const float4*)(p + 4);
  v8bf r;
  r[0] = (__bf16)x.x; r[1] = (__bf16)x.y; r[2] = (__bf16)x.z; r[3] = (__bf16)x.w;
  r[4] = (__bf16)y.x; r[5] = (__bf16)y.y; r[6] = (__bf16)y.z; r[7] = (__bf16)y.w;
  return r;
}

// --- gfx1250 async global->LDS copy (no VGPR round trip, tracked by ASYNCcnt)
// Generic pointers to LDS are (SHARED_BASE<<32 | lds_offset); low 32 bits are
// the wave-relative LDS byte address the instruction's VDST operand expects.
__device__ __forceinline__ unsigned lds_off32(const void* p) {
  return (unsigned)(unsigned long long)p;
}
__device__ __forceinline__ void async_ld_b128(unsigned ldsoff, const void* g) {
  asm volatile("global_load_async_to_lds_b128 %0, %1, off"
               :: "v"(ldsoff), "v"((unsigned long long)g)
               : "memory");
}

// ---------------------------------------------------------------------------
// Stage 0: transpose + convert weights to bf16, N-major.
// src: f32 [batch, rows, cols] -> dst: bf16 [batch, cols, rows]
// ---------------------------------------------------------------------------
__global__ void transpose_cvt_kernel(const float* __restrict__ src,
                                     __bf16* __restrict__ dst,
                                     int rows, int cols, int total) {
  int id = blockIdx.x * blockDim.x + threadIdx.x;
  if (id >= total) return;
  int rc = rows * cols;
  int b = id / rc;
  int rem = id - b * rc;
  int r = rem / cols;
  int c = rem - r * cols;
  dst[(size_t)b * rc + (size_t)c * rows + r] = (__bf16)src[id];
}

// ---------------------------------------------------------------------------
// Stage 1: per-head projection.  C[s, n] = sum_d X[b,s,d] * W[h,d,n]
// Wt is bf16 [H, 64, 1024] (N-major).  8 waves/block, wave tile = 16x64.
// STORET==0 : dst bf16 [B*H, S, 64] (row-major)       (Qi, Ki)
// STORET==1 : dst bf16 [B*H, 64, S] (transposed)      (ViT)
// ---------------------------------------------------------------------------
template <int STORET>
__global__ __launch_bounds__(256)
void proj_kernel(const float* __restrict__ X, const __bf16* __restrict__ Wt,
                 __bf16* __restrict__ dst) {
  const int lane = threadIdx.x & 31;
  const int wid  = threadIdx.x >> 5;
  const int l16  = lane & 15;
  const int kh   = lane >> 4;              // K-half selector (also C row half)
  const int bh   = blockIdx.y;             // b*H + h
  const int b    = bh >> 4;                // H_ == 16
  const int h    = bh & 15;
  const int sbase = blockIdx.x * 128;

  const float* arow = X + ((size_t)b * S_ + sbase + wid * 16 + l16) * D_;

  v8f acc[4] = {{}, {}, {}, {}};
  for (int k = 0; k < D_; k += 32) {
    FragU a;
    a.h[0] = cvt8(arow + k + kh * 8);
    a.h[1] = cvt8(arow + k + 16 + kh * 8);
#pragma unroll
    for (int n = 0; n < 4; ++n) {
      const __bf16* brow =
          Wt + ((size_t)h * 64 + n * 16 + l16) * D_ + k + kh * 8;
      acc[n] = wmma_bf16(a.v, load_frag(brow), acc[n]);
    }
  }

#pragma unroll
  for (int n = 0; n < 4; ++n) {
#pragma unroll
    for (int v = 0; v < 8; ++v) {
      int srow = sbase + wid * 16 + v + kh * 8;   // C row = v + 8*half
      int col  = n * 16 + l16;                    // C col = lane%16 + 16*n
      __bf16 val = (__bf16)acc[n][v];
      if (STORET == 0) dst[((size_t)bh * S_ + srow) * DK_ + col] = val;
      else             dst[((size_t)bh * DK_ + col) * S_ + srow] = val;
    }
  }
}

// ---------------------------------------------------------------------------
// Stage 2: flash attention.  One block = 128 Q rows of one (b,h); 8 waves,
// each wave owns 16 rows.  Keys processed 64 at a time, double-buffered
// through LDS with gfx1250 async global->LDS copies (ASYNCcnt completes in
// order, so s_wait_asynccnt 4 == "current tile landed, next still in flight").
// Row sums of P are computed on the matrix pipe with an all-ones B fragment;
// the row-max shuffles are batched 8-wide per xor-step so the ds_bpermute
// latency overlaps across independent rows.
// ---------------------------------------------------------------------------
__global__ __launch_bounds__(256)
void attn_kernel(const __bf16* __restrict__ Qi, const __bf16* __restrict__ Ki,
                 const __bf16* __restrict__ ViT, __bf16* __restrict__ Hd) {
  __shared__ __align__(16) __bf16 ldsK[2][64 * 64];    // [buf][key][dk]
  __shared__ __align__(16) __bf16 ldsV[2][64 * 64];    // [buf][dk][key]
  __shared__ __align__(16) __bf16 ldsP[8 * 16 * 64];   // wave-private P tiles

  const int lane = threadIdx.x & 31;
  const int wid  = threadIdx.x >> 5;
  const int l16  = lane & 15;
  const int kh   = lane >> 4;
  const int bh   = blockIdx.y;
  const int sbase = blockIdx.x * 128;

  const __bf16* Qih = Qi  + (size_t)bh * S_ * DK_;
  const __bf16* Kih = Ki  + (size_t)bh * S_ * DK_;
  const __bf16* Vih = ViT + (size_t)bh * DK_ * S_;

  // per-thread tile-copy coordinates (16B units; 512 units per 8KB tile)
  const int u0 = threadIdx.x;        // unit 0..255
  const int u1 = threadIdx.x + 256;  // unit 256..511

  // issue one K/V tile (4 async b128 loads per thread)
  auto stage = [&](int j0, int buf) {
    const v8bf* srcK = (const v8bf*)(Kih + (size_t)j0 * DK_);
    v8bf* dk = (v8bf*)ldsK[buf];
    async_ld_b128(lds_off32(dk + u0), srcK + u0);
    async_ld_b128(lds_off32(dk + u1), srcK + u1);
    v8bf* dv = (v8bf*)ldsV[buf];
    int row = u0 >> 3, cu = u0 & 7;
    async_ld_b128(lds_off32(dv + u0), Vih + (size_t)row * S_ + j0 + cu * 8);
    row = u1 >> 3; cu = u1 & 7;
    async_ld_b128(lds_off32(dv + u1), Vih + (size_t)row * S_ + j0 + cu * 8);
  };

  // Q fragments resident in registers: 16 rows x 64 dk -> 2 K-steps
  v16bf qf[2];
  {
    const __bf16* qrow = Qih + ((size_t)sbase + wid * 16 + l16) * DK_;
    qf[0] = load_frag(qrow + kh * 8);
    qf[1] = load_frag(qrow + 32 + kh * 8);
  }

  // all-ones B fragment: C[m][n] = sum_k A[m][k] for every n
  FragU onesU;
#pragma unroll
  for (int e = 0; e < 16; ++e) onesU.v[e] = (__bf16)1.0f;
  const v16bf onesf = onesU.v;

  v8f o[4] = {{}, {}, {}, {}};
  v8f lacc = {};               // running row sums (flash 'l' accumulator)
  float mi[8];
#pragma unroll
  for (int v = 0; v < 8; ++v) mi[v] = -1e30f;
  const float scale = 0.125f;  // 1/sqrt(64)

  const int NB = S_ / 64;      // 32 key blocks
  stage(0, 0);                 // prologue fill of buffer 0

  for (int i = 0; i < NB; ++i) {
    const __bf16* bk = ldsK[i & 1];
    const __bf16* bv = ldsV[i & 1];

    if (i + 1 < NB) {
      stage((i + 1) * 64, (i + 1) & 1);                 // prefetch next tile
      asm volatile("s_wait_asynccnt 4" ::: "memory");   // current tile done
    } else {
      asm volatile("s_wait_asynccnt 0" ::: "memory");
    }
    __syncthreads();   // all waves' pieces of the current tile are visible

    // S = Q x K^T  (16x64 scores per wave)
    v8f sacc[4] = {{}, {}, {}, {}};
#pragma unroll
    for (int kk = 0; kk < 2; ++kk) {
#pragma unroll
      for (int n = 0; n < 4; ++n) {
        v16bf kf = load_frag(&bk[(n * 16 + l16) * 64 + kk * 32 + kh * 8]);
        sacc[n] = wmma_bf16(qf[kk], kf, sacc[n]);
      }
    }
#pragma unroll
    for (int n = 0; n < 4; ++n)
#pragma unroll
      for (int v = 0; v < 8; ++v) sacc[n][v] *= scale;

    // ---- row max: batch all 8 rows per xor-distance so the 8 ds_bpermute
    // ---- per step issue back-to-back and share one wait.
    float mx[8];
#pragma unroll
    for (int v = 0; v < 8; ++v)
      mx[v] = fmaxf(fmaxf(sacc[0][v], sacc[1][v]),
                    fmaxf(sacc[2][v], sacc[3][v]));
#pragma unroll
    for (int d = 8; d >= 1; d >>= 1) {
      float t[8];
#pragma unroll
      for (int v = 0; v < 8; ++v) t[v] = __shfl_xor(mx[v], d, 32);
#pragma unroll
      for (int v = 0; v < 8; ++v) mx[v] = fmaxf(mx[v], t[v]);
    }

    // online softmax update; row (v + 8*kh) lives in one 16-lane half
    float alpha[8];
#pragma unroll
    for (int v = 0; v < 8; ++v) {
      float mnew = fmaxf(mi[v], mx[v]);
      alpha[v] = __expf(mi[v] - mnew);
#pragma unroll
      for (int n = 0; n < 4; ++n)
        sacc[n][v] = __expf(sacc[n][v] - mnew);
      mi[v] = mnew;
    }

    // C-layout -> A-layout for P via wave-private LDS; rescale O and L by alpha
    __bf16* P = &ldsP[wid * 16 * 64];
#pragma unroll
    for (int n = 0; n < 4; ++n) {
#pragma unroll
      for (int v = 0; v < 8; ++v) {
        P[(v + kh * 8) * 64 + n * 16 + l16] = (__bf16)sacc[n][v];
        o[n][v] *= alpha[v];
      }
    }
#pragma unroll
    for (int v = 0; v < 8; ++v) lacc[v] *= alpha[v];

    // same-wave LDS ops are in-order; compiler inserts s_wait_dscnt
#pragma unroll
    for (int kk = 0; kk < 2; ++kk) {
      v16bf pf = load_frag(&P[l16 * 64 + kk * 32 + kh * 8]);
      lacc = wmma_bf16(pf, onesf, lacc);   // row sums on the matrix pipe
#pragma unroll
      for (int n = 0; n < 4; ++n) {
        v16bf vf = load_frag(&bv[(n * 16 + l16) * 64 + kk * 32 + kh * 8]);
        o[n] = wmma_bf16(pf, vf, o[n]);
      }
    }
    __syncthreads();   // everyone done with this buffer before it is refilled
  }

  // epilogue: divide by row sums (per-lane, no broadcast needed), store bf16
#pragma unroll
  for (int n = 0; n < 4; ++n) {
#pragma unroll
    for (int v = 0; v < 8; ++v) {
      int srow = sbase + wid * 16 + v + kh * 8;
      Hd[((size_t)bh * S_ + srow) * DK_ + n * 16 + l16] =
          (__bf16)(o[n][v] / lacc[v]);
    }
  }
}

// ---------------------------------------------------------------------------
// Stage 3: output projection.  The reference's raw reshape makes the head
// buffer exactly a contiguous [B*S, 1024] matrix.  out = X @ WO  (fp32 out).
// ---------------------------------------------------------------------------
__global__ __launch_bounds__(256)
void outproj_kernel(const __bf16* __restrict__ Xb, const __bf16* __restrict__ WOt,
                    float* __restrict__ out) {
  const int lane = threadIdx.x & 31;
  const int wid  = threadIdx.x >> 5;
  const int l16  = lane & 15;
  const int kh   = lane >> 4;
  const int mbase = blockIdx.x * 128;   // over B*S = 4096
  const int nbase = blockIdx.y * 64;    // over DMODEL = 1024

  const __bf16* arow = Xb + ((size_t)mbase + wid * 16 + l16) * D_;
  v8f acc[4] = {{}, {}, {}, {}};
  for (int k = 0; k < D_; k += 32) {
    v16bf a = load_frag(arow + k + kh * 8);
#pragma unroll
    for (int n = 0; n < 4; ++n) {
      v16bf b = load_frag(WOt + ((size_t)nbase + n * 16 + l16) * D_ + k + kh * 8);
      acc[n] = wmma_bf16(a, b, acc[n]);
    }
  }
#pragma unroll
  for (int n = 0; n < 4; ++n) {
#pragma unroll
    for (int v = 0; v < 8; ++v) {
      int r = mbase + wid * 16 + v + kh * 8;
      out[(size_t)r * D_ + nbase + n * 16 + l16] = acc[n][v];
    }
  }
}

// ---------------------------------------------------------------------------
extern "C" void kernel_launch(void* const* d_in, const int* in_sizes, int n_in,
                              void* d_out, int out_size, void* d_ws, size_t ws_size,
                              hipStream_t stream) {
  const float* Q  = (const float*)d_in[0];
  const float* K  = (const float*)d_in[1];
  const float* V  = (const float*)d_in[2];
  const float* WQ = (const float*)d_in[3];
  const float* WK = (const float*)d_in[4];
  const float* WV = (const float*)d_in[5];
  const float* WO = (const float*)d_in[6];
  float* out = (float*)d_out;

  // Workspace layout (bf16 elements): 4x 1M weights + 4x 4M activations = 40MB
  __bf16* WQt = (__bf16*)d_ws;
  __bf16* WKt = WQt + (1u << 20);
  __bf16* WVt = WKt + (1u << 20);
  __bf16* WOt = WVt + (1u << 20);
  __bf16* Qi  = WOt + (1u << 20);
  __bf16* Ki  = Qi  + (4u << 20);
  __bf16* ViT = Ki  + (4u << 20);
  __bf16* Hd  = ViT + (4u << 20);

  // Stage 0: weight transpose+convert
  {
    int total = H_ * D_ * DK_;     // 1M
    int nb = (total + 255) / 256;
    transpose_cvt_kernel<<<nb, 256, 0, stream>>>(WQ, WQt, D_, DK_, total);
    transpose_cvt_kernel<<<nb, 256, 0, stream>>>(WK, WKt, D_, DK_, total);
    transpose_cvt_kernel<<<nb, 256, 0, stream>>>(WV, WVt, D_, DK_, total);
    int t2 = D_ * D_;
    transpose_cvt_kernel<<<(t2 + 255) / 256, 256, 0, stream>>>(WO, WOt, D_, D_, t2);
  }

  // Stage 1: projections (V stored transposed for the PV GEMM)
  dim3 pg(S_ / 128, B_ * H_);
  proj_kernel<0><<<pg, 256, 0, stream>>>(Q, WQt, Qi);
  proj_kernel<0><<<pg, 256, 0, stream>>>(K, WKt, Ki);
  proj_kernel<1><<<pg, 256, 0, stream>>>(V, WVt, ViT);

  // Stage 2: flash attention
  dim3 ag(S_ / 128, B_ * H_);
  attn_kernel<<<ag, 256, 0, stream>>>(Qi, Ki, ViT, Hd);

  // Stage 3: output projection
  dim3 og((B_ * S_) / 128, D_ / 64);
  outproj_kernel<<<og, 256, 0, stream>>>(Hd, WOt, out);
}
